// GConvLSTM_36661840838809
// MI455X (gfx1250) — compile-verified, matrix operand order checked by reading, not run on hardware
//
#include <hip/hip_runtime.h>
#include <hip/hip_bf16.h>

typedef __attribute__((ext_vector_type(16))) _Float16 v16h;
typedef __attribute__((ext_vector_type(8)))  float    v8f;

// Problem constants
#define BB     4
#define TT     16
#define HID    32
#define HW     4096           // 64*64
#define CH_ALL 288            // (CIN + 2*HID) * 4 rotations
#define CH_XH  160            // (CIN + HID) * 4
#define LCOLS  66             // 64 + 2 halo columns
#define LCH    198            // 3 rows * 66 cols per channel in LDS

// ---------------------------------------------------------------------------
// Weight transform: build rotated P4 weights in WMMA A-fragment order (f16).
// K ordering: k = (kh*3+kw)*Ci4 + ic4.  Fragment layout (16-bit A, wave32):
//   m   = mt*16 + (lane & 15)
//   k   = kt*32 + koff,  koff = e + (e>=8 ? 8:0) + (lane>=16 ? 8:0)
// ---------------------------------------------------------------------------
__global__ void wt_transform(const float* __restrict__ w, _Float16* __restrict__ wfrag,
                             int Ci, int MTILES, int SDIV) {
    const int KSTEPS = 9 * SDIV;
    const int total = MTILES * KSTEPS * 512;
    int idx = blockIdx.x * blockDim.x + threadIdx.x;
    if (idx >= total) return;
    const int e    = idx & 15;
    const int lane = (idx >> 4) & 31;
    const int kt   = (idx >> 9) % KSTEPS;
    const int mt   = (idx >> 9) / KSTEPS;
    const int koff = e + ((e >= 8) ? 8 : 0) + ((lane >= 16) ? 8 : 0);
    const int m    = mt * 16 + (lane & 15);       // oc4
    const int s    = kt / SDIV;                   // 3x3 tap index
    const int ic4  = (kt % SDIV) * 32 + koff;
    const int o = m >> 2, r = m & 3;
    const int i = ic4 >> 2, si = ic4 & 3;
    int a = s / 3, b2 = s % 3;
    for (int rr = 0; rr < r; ++rr) { int ta = a; a = b2; b2 = 2 - ta; }  // rot90^r index map
    const int s2 = (si - r) & 3;                  // roll over rotation axis
    const float val = w[(((size_t)o * Ci + i) * 4 + s2) * 9 + a * 3 + b2];
    wfrag[idx] = (_Float16)val;
}

// ---------------------------------------------------------------------------
// Implicit-GEMM P4 conv via v_wmma_f32_16x16x32_f16, LDS-staged activations.
// Block: 128 threads = 4 waves, covering one image row (64 pixels); each wave
// owns a distinct 16-pixel N-tile and ALL M-tiles.  The 3x66 halo-padded patch
// of all CI4 channels is staged once into LDS (zero-filled halo -> branchless
// inner gather via immediate-offset DS loads).  Grid: (64 rows, 4 batches).
// ---------------------------------------------------------------------------
template<int MTILES, int SDIV>
__global__ __launch_bounds__(128)
void gconv_wmma(const _Float16* __restrict__ xpack,   // [B][288][64][64] f16
                const _Float16* __restrict__ wfrag,   // [MTILES][KSTEPS][32][16] f16
                const float*    __restrict__ bias,    // [MTILES*16/4]
                float*          __restrict__ out)     // [B][MTILES*16][4096] f32
{
    constexpr int KSTEPS = 9 * SDIV;
    constexpr int CI4    = SDIV * 32;
    extern __shared__ _Float16 lds[];                 // [CI4][3][66]

    const int lane = threadIdx.x & 31;
    const int wave = threadIdx.x >> 5;
    const int b    = blockIdx.y;
    const int row  = blockIdx.x;                      // output image row
    const int wpix = wave * 16 + (lane & 15);         // column within the row
    const int n    = row * 64 + wpix;                 // pixel (N index)
    const int kadd = (lane >= 16) ? 8 : 0;

    // ---- Stage input patch: CI4 channels x 3 rows x 66 cols, zero halo ----
    const _Float16* xb_base = xpack + (size_t)b * CH_ALL * HW;
    for (int i = threadIdx.x; i < CI4 * LCH; i += 128) {
        const int ch = i / LCH;
        const int rm = i % LCH;
        const int rr = rm / LCOLS;
        const int cc = rm % LCOLS;
        const int hh = row + rr - 1;
        const int ww = cc - 1;
        _Float16 v = (_Float16)0.0f;
        if (hh >= 0 && hh < 64 && ww >= 0 && ww < 64)
            v = xb_base[(size_t)ch * HW + hh * 64 + ww];
        lds[i] = v;
    }
    __syncthreads();

    v8f acc[MTILES];
    #pragma unroll
    for (int mt = 0; mt < MTILES; ++mt)
        #pragma unroll
        for (int j = 0; j < 8; ++j) acc[mt][j] = 0.0f;

    for (int s = 0; s < 9; ++s) {
        const int dh = s / 3, dw = s % 3;             // tap (0..2, 0..2)
        const int ptap = dh * LCOLS + wpix + dw;      // halo-adjusted base in LDS
        for (int kc = 0; kc < SDIV; ++kc) {
            const int kt = s * SDIV + kc;
            // Branchless B-fragment gather from LDS (K=32 x N=16 f16 layout).
            const _Float16* lb = lds + (size_t)(kc * 32 + kadd) * LCH + ptap;
            v16h bf;
            #pragma unroll
            for (int e = 0; e < 16; ++e) {
                const int ko = e + ((e >= 8) ? 8 : 0);
                bf[e] = lb[ko * LCH];
            }
            const _Float16* wrow = wfrag + ((size_t)kt * 32 + lane) * 16;
            // Prefetch next K-step's weight fragments (global_prefetch_b8).
            __builtin_prefetch(wrow + (size_t)32 * 16, 0, 3);
            #pragma unroll
            for (int mt = 0; mt < MTILES; ++mt) {
                const v16h af = *(const v16h*)(wrow + (size_t)mt * KSTEPS * 512);
                acc[mt] = __builtin_amdgcn_wmma_f32_16x16x32_f16(
                    false, af, false, bf, (short)0, acc[mt], false, false);
            }
        }
    }

    // Store C (16x16 f32, 8 VGPRs: lanes<16 -> M=j, lanes>=16 -> M=j+8) + bias.
    #pragma unroll
    for (int mt = 0; mt < MTILES; ++mt) {
        #pragma unroll
        for (int j = 0; j < 8; ++j) {
            const int m = mt * 16 + j + ((lane >= 16) ? 8 : 0);
            out[((size_t)b * (MTILES * 16) + m) * HW + n] = acc[mt][j] + bias[m >> 2];
        }
    }
}

// ---------------------------------------------------------------------------
// Elementwise kernels
// ---------------------------------------------------------------------------
__device__ __forceinline__ float sigm(float x) { return 1.0f / (1.0f + __expf(-x)); }

__global__ void init_state(float* h_state, float* c_state, _Float16* xpack) {
    int idx = blockIdx.x * blockDim.x + threadIdx.x;     // < 4*256*4096
    if (idx >= BB * 256 * HW) return;
    const int b = idx >> 20;             // 256*4096 = 2^20
    const int rem = idx & 1048575;
    if (rem < 524288) h_state[((size_t)b << 19) + rem] = 0.0f;
    else              c_state[((size_t)b << 19) + (rem - 524288)] = 0.0f;
    xpack[(size_t)b * CH_ALL * HW + (size_t)32 * HW + rem] = (_Float16)0.0f;  // h|c slots
}

__global__ void pack_x(const float* __restrict__ xin, _Float16* __restrict__ xpack, int t) {
    int idx = blockIdx.x * blockDim.x + threadIdx.x;     // < 4*32*4096
    if (idx >= BB * 32 * HW) return;
    const int b = idx >> 17;             // 32*4096 = 2^17
    const int rem = idx & 131071;
    const float v = xin[(((size_t)b * TT + t) << 17) + rem];
    xpack[(size_t)b * CH_ALL * HW + rem] = (_Float16)v;
}

__global__ void gate_kernel(const float* __restrict__ cc, const float* __restrict__ gg,
                            float* __restrict__ c_state, _Float16* __restrict__ xpack) {
    int idx = blockIdx.x * blockDim.x + threadIdx.x;     // < 4*128*4096
    if (idx >= BB * 128 * HW) return;
    const int b = idx >> 19;             // 128*4096 = 2^19
    const int rem = idx & 524287;
    const float iv = cc[(size_t)b * 256 * HW + rem];
    const float fv = cc[(size_t)b * 256 * HW + (size_t)128 * HW + rem];
    const float gv = gg[idx];
    const float cv = c_state[idx];
    const float cn = sigm(fv) * cv + sigm(iv) * tanhf(gv);
    c_state[idx] = cn;
    xpack[(size_t)b * CH_ALL * HW + (size_t)CH_XH * HW + rem] = (_Float16)cn;  // c slot
}

__global__ void out_kernel(const float* __restrict__ oo, const float* __restrict__ c_state,
                           float* __restrict__ h_state, _Float16* __restrict__ xpack,
                           float* __restrict__ layer_out, int t) {
    int idx = blockIdx.x * blockDim.x + threadIdx.x;     // < 4*128*4096
    if (idx >= BB * 128 * HW) return;
    const int b = idx >> 19;
    const int rem = idx & 524287;
    const float hn = sigm(oo[idx]) * tanhf(c_state[idx]);
    h_state[idx] = hn;
    xpack[(size_t)b * CH_ALL * HW + (size_t)32 * HW + rem] = (_Float16)hn;     // h slot
    layer_out[(((size_t)b * TT + t) << 19) + rem] = hn;
}

__global__ void copy_last(const float* __restrict__ h_state, const float* __restrict__ c_state,
                          float* __restrict__ out_tail) {
    int idx = blockIdx.x * blockDim.x + threadIdx.x;     // < 2*2097152
    if (idx >= 2 * BB * 128 * HW) return;
    if (idx < BB * 128 * HW) out_tail[idx] = h_state[idx];
    else                     out_tail[idx] = c_state[idx - BB * 128 * HW];
}

// ---------------------------------------------------------------------------
extern "C" void kernel_launch(void* const* d_in, const int* in_sizes, int n_in,
                              void* d_out, int out_size, void* d_ws, size_t ws_size,
                              hipStream_t stream) {
    const float* x   = (const float*)d_in[0];
    const float* wfi = (const float*)d_in[1];
    const float* bfi = (const float*)d_in[2];
    const float* wg  = (const float*)d_in[3];
    const float* bg  = (const float*)d_in[4];
    const float* wo  = (const float*)d_in[5];
    const float* bo  = (const float*)d_in[6];
    float* out = (float*)d_out;

    // Workspace carve-up (256B aligned)
    char* ws = (char*)d_ws;
    size_t off = 0;
    auto alloc = [&](size_t bytes) -> void* {
        void* p = ws + off; off += (bytes + 255) & ~(size_t)255; return p;
    };
    _Float16* wfrag_fi = (_Float16*)alloc((size_t)16 * 81 * 512 * 2);   // 1.27 MB
    _Float16* wfrag_g  = (_Float16*)alloc((size_t)8  * 45 * 512 * 2);   // 0.35 MB
    _Float16* wfrag_o  = (_Float16*)alloc((size_t)8  * 81 * 512 * 2);   // 0.63 MB
    _Float16* xpack    = (_Float16*)alloc((size_t)BB * CH_ALL * HW * 2);// 9.4 MB [x|h|c]
    float* h_state = (float*)alloc((size_t)BB * 128 * HW * 4);          // 8.4 MB
    float* c_state = (float*)alloc((size_t)BB * 128 * HW * 4);          // 8.4 MB
    float* cc      = (float*)alloc((size_t)BB * 256 * HW * 4);          // 16.8 MB
    float* gg      = (float*)alloc((size_t)BB * 128 * HW * 4);          // 8.4 MB
    float* oo      = (float*)alloc((size_t)BB * 128 * HW * 4);          // 8.4 MB

    // One-time weight transforms into WMMA fragment order.
    wt_transform<<<(16 * 81 * 512 + 255) / 256, 256, 0, stream>>>(wfi, wfrag_fi, 72, 16, 9);
    wt_transform<<<(8  * 45 * 512 + 255) / 256, 256, 0, stream>>>(wg,  wfrag_g,  40, 8,  5);
    wt_transform<<<(8  * 81 * 512 + 255) / 256, 256, 0, stream>>>(wo,  wfrag_o,  72, 8,  9);
    init_state<<<(BB * 256 * HW) / 256, 256, 0, stream>>>(h_state, c_state, xpack);

    const dim3 cgrid(64, BB);                     // one image row per block
    const size_t lds_288 = (size_t)288 * LCH * 2; // 114,048 B
    const size_t lds_160 = (size_t)160 * LCH * 2; //  63,360 B
    for (int t = 0; t < TT; ++t) {
        pack_x<<<(BB * 32 * HW) / 256, 256, 0, stream>>>(x, xpack, t);
        // cc = p4conv(xhc, w_fi)  : 288 -> 256
        gconv_wmma<16, 9><<<cgrid, 128, lds_288, stream>>>(xpack, wfrag_fi, bfi, cc);
        // gg = p4conv(xh, w_g)    : 160 -> 128
        gconv_wmma<8, 5><<<cgrid, 128, lds_160, stream>>>(xpack, wfrag_g, bg, gg);
        // c_next = sigmoid(f)*c + sigmoid(i)*tanh(gg)
        gate_kernel<<<(BB * 128 * HW) / 256, 256, 0, stream>>>(cc, gg, c_state, xpack);
        // oo = p4conv([xh|c_next], w_o) : 288 -> 128
        gconv_wmma<8, 9><<<cgrid, 128, lds_288, stream>>>(xpack, wfrag_o, bo, oo);
        // h_next = sigmoid(oo)*tanh(c_next); write layer_output[:, t]
        out_kernel<<<(BB * 128 * HW) / 256, 256, 0, stream>>>(oo, c_state, h_state, xpack, out, t);
    }
    // Append h_last, c_last after layer_output.
    copy_last<<<(2 * BB * 128 * HW) / 256, 256, 0, stream>>>(
        h_state, c_state, out + (size_t)BB * TT * 128 * HW);
}
